// DeeppH_49417893708237
// MI455X (gfx1250) — matrix operand order repaired and checked by reading, not compiled, not run on hardware
//
#include <hip/hip_runtime.h>
#include <math.h>

// Problem constants (match reference)
#define NN    131072
#define BNUM  64
#define HID   512
#define SEQ   1024
#define TWOH  1024
#define DENSE 16
#define HEADS 4
#define NT    4     // N tiles per wave (4 x 16 = 64-wide strip)
#define AP    36    // padded f32 row stride in LDS (16B-aligned, bank-spread)

typedef __attribute__((ext_vector_type(16))) __bf16 v16bf;
typedef __attribute__((ext_vector_type(8)))  float  v8f;

__device__ __forceinline__ v8f wmma_bf16(v16bf a, v16bf b, v8f c) {
  // D = A(16x32 bf16) x B(32x16 bf16) + C(16x16 f32)
  return __builtin_amdgcn_wmma_f32_16x16x32_bf16(false, a, false, b, (short)0, c,
                                                 false, false);
}

__device__ __forceinline__ void wait_async0() {
#if __has_builtin(__builtin_amdgcn_s_wait_asynccnt)
  __builtin_amdgcn_s_wait_asynccnt(0);
#else
  asm volatile("s_wait_asynccnt 0x0" ::: "memory");
#endif
}

// f32 -> bf16 hi + bf16 lo  (x ~= hi + lo); bf16x3 GEMM gives ~fp32 accuracy
__device__ __forceinline__ void bsplit(float x, __bf16 &h, __bf16 &l) {
  h = (__bf16)x;
  l = (__bf16)(x - (float)h);
}

// Build the 16-bit A fragment for one K=32 chunk from an
// f32 row pointer (global or LDS). lanes0-15: elems0-15 = K+0..15;
// lanes16-31: K+16..31.
__device__ __forceinline__ void load_a_frag(const float* row,
                                            int half, v16bf &hi, v16bf &lo) {
  const float* p = row + half * 16;
  float4 a0 = *(const float4*)(p);
  float4 a1 = *(const float4*)(p + 4);
  float4 a2 = *(const float4*)(p + 8);
  float4 a3 = *(const float4*)(p + 12);
  float av[16] = {a0.x,a0.y,a0.z,a0.w, a1.x,a1.y,a1.z,a1.w,
                  a2.x,a2.y,a2.z,a2.w, a3.x,a3.y,a3.z,a3.w};
#pragma unroll
  for (int i = 0; i < 16; ++i) { __bf16 h, l; bsplit(av[i], h, l); hi[i]=h; lo[i]=l; }
}

// ---- weight prep: split + transpose to K-contiguous bf16 hi/lo --------------
__global__ void k_split_wproj(const float* __restrict__ W, __bf16* Wh, __bf16* Wl) {
  int idx = blockIdx.x * blockDim.x + threadIdx.x;
  if (idx >= SEQ * HID) return;
  int k = idx / HID, c = idx % HID;
  __bf16 h, l; bsplit(W[idx], h, l);
  Wh[(size_t)c * SEQ + k] = h;
  Wl[(size_t)c * SEQ + k] = l;
}
__global__ void k_split_wa1(const float* __restrict__ W, __bf16* Wh, __bf16* Wl) {
  int idx = blockIdx.x * blockDim.x + threadIdx.x;
  if (idx >= TWOH * DENSE) return;
  int k = idx / DENSE, c = idx % DENSE;
  __bf16 h, l; bsplit(W[idx], h, l);
  Wh[(size_t)c * TWOH + k] = h;
  Wl[(size_t)c * TWOH + k] = l;
}

// ---- K1: proj = seq_feat @ W_proj + b_proj ----------------------------------
// Block = 32 rows x 512 cols; wave = 2 M-tiles x 4 N-tiles. A chunk (32x32 f32)
// staged into LDS with GLOBAL_LOAD_ASYNC_TO_LDS_B128, double-buffered so the
// DMA of chunk c+1 overlaps the 24 WMMAs of chunk c. The bf16x3 passes are
// emitted as sweeps over independent accumulators so adjacent WMMAs never
// share an accumulator (no D->A/B hazards, no v_nop trains).
__global__ void __launch_bounds__(256) k_proj_gemm(
    const float* __restrict__ seq, const __bf16* __restrict__ Wh,
    const __bf16* __restrict__ Wl, const float* __restrict__ bias,
    float* __restrict__ proj) {
  __shared__ float sA[2][32 * AP];

  const int tid  = threadIdx.x;
  const int lane = tid & 31;
  const int wave = tid >> 5;               // 0..7 -> N strip
  const int mblk = blockIdx.x * 32;
  const int n0   = wave * (16 * NT);       // 64-wide strip
  const int half = lane >> 4, l15 = lane & 15;

  // staging map: 256 threads x 16B cover the 32x32 f32 chunk
  const int srow = tid >> 3;               // 0..31
  const int scol = (tid & 7) * 4;          // 0,4,...,28
  const float* gsrc = seq + (size_t)(mblk + srow) * SEQ + scol;
  const unsigned lbase = (unsigned)((srow * AP + scol) * 4);   // LDS byte offset

  v8f acc0[NT] = {};   // M tile 0 (rows mblk .. mblk+15)
  v8f acc1[NT] = {};   // M tile 1 (rows mblk+16 .. mblk+31)

  { // prologue: async-stage chunk 0 into buffer 0
    unsigned loff = lbase;
    unsigned long long ga = (unsigned long long)(uintptr_t)gsrc;
    asm volatile("global_load_async_to_lds_b128 %0, %1, off"
                 :: "v"(loff), "v"(ga) : "memory");
  }

  for (int c = 0; c < 32; ++c) {
    const int k0 = c * 32;
    wait_async0();            // only chunk c's copy is outstanding here
    __syncthreads();          // chunk c visible to all waves; prior reads done
    if (c + 1 < 32) {         // overlap: stage chunk c+1 into the other buffer
      unsigned loff = lbase + (unsigned)(((c + 1) & 1) * 32 * AP * 4);
      unsigned long long ga = (unsigned long long)(uintptr_t)(gsrc + k0 + 32);
      asm volatile("global_load_async_to_lds_b128 %0, %1, off"
                   :: "v"(loff), "v"(ga) : "memory");
    }
    // A fragments for both M tiles (split to bf16 hi/lo from LDS)
    v16bf ah0, al0, ah1, al1;
    load_a_frag(&sA[c & 1][(l15)      * AP], half, ah0, al0);
    load_a_frag(&sA[c & 1][(16 + l15) * AP], half, ah1, al1);
    // B fragments: lane = column n, 16 contiguous K values
    v16bf bh[NT], bl[NT];
#pragma unroll
    for (int t = 0; t < NT; ++t) {
      const size_t bo = (size_t)(n0 + t * 16 + l15) * SEQ + k0 + half * 16;
      bh[t] = *(const v16bf*)(Wh + bo);
      bl[t] = *(const v16bf*)(Wl + bo);
    }
    // six sweeps of 4 independent WMMAs; same-acc reuse distance = 8
#pragma unroll
    for (int t = 0; t < NT; ++t) acc0[t] = wmma_bf16(ah0, bh[t], acc0[t]);
#pragma unroll
    for (int t = 0; t < NT; ++t) acc1[t] = wmma_bf16(ah1, bh[t], acc1[t]);
#pragma unroll
    for (int t = 0; t < NT; ++t) acc0[t] = wmma_bf16(al0, bh[t], acc0[t]);
#pragma unroll
    for (int t = 0; t < NT; ++t) acc1[t] = wmma_bf16(al1, bh[t], acc1[t]);
#pragma unroll
    for (int t = 0; t < NT; ++t) acc0[t] = wmma_bf16(ah0, bl[t], acc0[t]);
#pragma unroll
    for (int t = 0; t < NT; ++t) acc1[t] = wmma_bf16(ah1, bl[t], acc1[t]);
  }
#pragma unroll
  for (int t = 0; t < NT; ++t) {
    const int nc = n0 + t * 16 + l15;
    const float b = bias[nc];
#pragma unroll
    for (int r = 0; r < 8; ++r) {          // C layout: vgpr r, halves -> M
      proj[(size_t)(mblk      + r + half * 8) * HID + nc] = acc0[t][r] + b;
      proj[(size_t)(mblk + 16 + r + half * 8) * HID + nc] = acc1[t][r] + b;
    }
  }
}

// ---- K2: scores = tanh(feat@W_a1 + b_a1) @ W_a2 + b_a2 ----------------------
__global__ void __launch_bounds__(256) k_scores(
    const float* __restrict__ structf, const float* __restrict__ proj,
    const __bf16* __restrict__ A1h, const __bf16* __restrict__ A1l,
    const float* __restrict__ b_a1, const float* __restrict__ W_a2,
    const float* __restrict__ b_a2, float* __restrict__ scores) {
  const int lane = threadIdx.x & 31;
  const int wave = blockIdx.x * 8 + (threadIdx.x >> 5);
  const int m0 = wave * 16;
  const int half = lane >> 4, l15 = lane & 15;

  v8f acc = {};
  for (int k0 = 0; k0 < TWOH; k0 += 32) {
    const float* src = (k0 < HID)
        ? (structf + (size_t)(m0 + l15) * HID + k0)
        : (proj    + (size_t)(m0 + l15) * HID + (k0 - HID));
    v16bf ah, al;
    load_a_frag(src, half, ah, al);
    const size_t bo = (size_t)l15 * TWOH + k0 + half * 16;
    v16bf bh = *(const v16bf*)(A1h + bo);
    v16bf bl = *(const v16bf*)(A1l + bo);
    acc = wmma_bf16(ah, bh, acc);
    acc = wmma_bf16(al, bh, acc);
    acc = wmma_bf16(ah, bl, acc);
  }
  float hidden[8];
  const float ba = b_a1[l15];
#pragma unroll
  for (int r = 0; r < 8; ++r) hidden[r] = tanhf(acc[r] + ba);
  // tiny 16x4 matmul: reduce over 16 C-layout columns (lanes within each half)
#pragma unroll
  for (int h = 0; h < HEADS; ++h) {
    const float w = W_a2[l15 * HEADS + h];
#pragma unroll
    for (int r = 0; r < 8; ++r) {
      float v = hidden[r] * w;
      v += __shfl_xor(v, 1); v += __shfl_xor(v, 2);
      v += __shfl_xor(v, 4); v += __shfl_xor(v, 8);
      if (l15 == 0)
        scores[(size_t)(m0 + r + half * 8) * HEADS + h] = v + b_a2[h];
    }
  }
}

// ---- segment machinery (batch_id is sorted -> binary-search boundaries) -----
__global__ void k_bounds(const int* __restrict__ bid, int* __restrict__ segstart) {
  int g = blockIdx.x * blockDim.x + threadIdx.x;
  if (g > BNUM) return;
  int lo = 0, hi = NN;
  while (lo < hi) { int mid = (lo + hi) >> 1; if (bid[mid] < g) lo = mid + 1; else hi = mid; }
  segstart[g] = lo;
}

__global__ void __launch_bounds__(256) k_seg_stats(
    const float4* __restrict__ scores4, const int* __restrict__ segstart,
    float4* __restrict__ segm, float4* __restrict__ segd) {
  __shared__ float4 red[256];
  const int g = blockIdx.x, tid = threadIdx.x;
  const int s = segstart[g], e = segstart[g + 1];
  float4 mx = make_float4(-3.0e38f, -3.0e38f, -3.0e38f, -3.0e38f);
  for (int i = s + tid; i < e; i += 256) {
    float4 sc = scores4[i];
    mx.x = fmaxf(mx.x, sc.x); mx.y = fmaxf(mx.y, sc.y);
    mx.z = fmaxf(mx.z, sc.z); mx.w = fmaxf(mx.w, sc.w);
  }
  red[tid] = mx; __syncthreads();
  for (int o = 128; o > 0; o >>= 1) {
    if (tid < o) {
      float4 a = red[tid], b = red[tid + o];
      a.x = fmaxf(a.x, b.x); a.y = fmaxf(a.y, b.y);
      a.z = fmaxf(a.z, b.z); a.w = fmaxf(a.w, b.w);
      red[tid] = a;
    }
    __syncthreads();
  }
  const float4 m = red[0]; __syncthreads();
  float4 sm = make_float4(0.f, 0.f, 0.f, 0.f);
  for (int i = s + tid; i < e; i += 256) {
    float4 sc = scores4[i];
    sm.x += expf(sc.x - m.x); sm.y += expf(sc.y - m.y);
    sm.z += expf(sc.z - m.z); sm.w += expf(sc.w - m.w);
  }
  red[tid] = sm; __syncthreads();
  for (int o = 128; o > 0; o >>= 1) {
    if (tid < o) {
      float4 a = red[tid], b = red[tid + o];
      a.x += b.x; a.y += b.y; a.z += b.z; a.w += b.w;
      red[tid] = a;
    }
    __syncthreads();
  }
  if (tid == 0) { segm[g] = m; segd[g] = red[0]; }
}

__global__ void k_attn(const float4* __restrict__ scores4, const int* __restrict__ bid,
                       const float4* __restrict__ segm, const float4* __restrict__ segd,
                       float4* __restrict__ attn_out, float* __restrict__ asum) {
  int i = blockIdx.x * blockDim.x + threadIdx.x;
  if (i >= NN) return;
  int g = bid[i];
  float4 m = segm[g], d = segd[g], sc = scores4[i];
  float4 a;
  a.x = expf(sc.x - m.x) / d.x; a.y = expf(sc.y - m.y) / d.y;
  a.z = expf(sc.z - m.z) / d.z; a.w = expf(sc.w - m.w) / d.w;
  attn_out[i] = a;
  asum[i] = a.x + a.y + a.z + a.w;
}

__global__ void __launch_bounds__(256) k_pooled(
    const float* __restrict__ structf, const float* __restrict__ proj,
    const float* __restrict__ asum, const int* __restrict__ segstart,
    float* __restrict__ pooled) {
  const int g = blockIdx.x, tid = threadIdx.x;
  const int s = segstart[g], e = segstart[g + 1];
  float a0 = 0.f, a1 = 0.f, a2 = 0.f, a3 = 0.f;
  for (int i = s; i < e; ++i) {
    const float w = asum[i];
    const float* sr = structf + (size_t)i * HID;
    const float* pr = proj    + (size_t)i * HID;
    a0 += sr[tid]       * w;
    a1 += sr[tid + 256] * w;
    a2 += pr[tid]       * w;
    a3 += pr[tid + 256] * w;
  }
  float* dst = pooled + (size_t)g * TWOH;
  dst[tid] = a0; dst[tid + 256] = a1; dst[tid + 512] = a2; dst[tid + 768] = a3;
}

__global__ void __launch_bounds__(256) k_emb(
    const float* __restrict__ pooled, const float* __restrict__ W_f1,
    const float* __restrict__ b_f1, float* __restrict__ emb) {
  __shared__ float prow[TWOH];
  const int g = blockIdx.x, tid = threadIdx.x;
  for (int j = tid; j < TWOH; j += 256) prow[j] = pooled[(size_t)g * TWOH + j];
  __syncthreads();
  for (int jj = tid; jj < HID; jj += 256) {
    float s = b_f1[jj];
    for (int k = 0; k < TWOH; ++k) s += prow[k] * W_f1[(size_t)k * HID + jj];
    emb[(size_t)g * HID + jj] = (s > 0.f) ? s : (expf(s) - 1.f);   // elu
  }
}

__global__ void k_out(const float* __restrict__ emb, const float* __restrict__ W_f2,
                      const float* __restrict__ b_f2, float* __restrict__ out) {
  int t = threadIdx.x;
  if (t >= BNUM * 2) return;
  int g = t >> 1, k = t & 1;
  float s = b_f2[k];
  for (int c = 0; c < HID; ++c) s += emb[(size_t)g * HID + c] * W_f2[c * 2 + k];
  out[t] = s;
}

extern "C" void kernel_launch(void* const* d_in, const int* in_sizes, int n_in,
                              void* d_out, int out_size, void* d_ws, size_t ws_size,
                              hipStream_t stream) {
  (void)in_sizes; (void)n_in; (void)out_size; (void)ws_size;
  const float* structure = (const float*)d_in[0];
  const float* seq       = (const float*)d_in[1];
  const int*   bid       = (const int*)d_in[2];
  const float* W_proj    = (const float*)d_in[4];
  const float* b_proj    = (const float*)d_in[5];
  const float* W_a1      = (const float*)d_in[6];
  const float* b_a1      = (const float*)d_in[7];
  const float* W_a2      = (const float*)d_in[8];
  const float* b_a2      = (const float*)d_in[9];
  const float* W_f1      = (const float*)d_in[10];
  const float* b_f1      = (const float*)d_in[11];
  const float* W_f2      = (const float*)d_in[12];
  const float* b_f2      = (const float*)d_in[13];
  float* out = (float*)d_out;

  // Workspace carve-up (~277 MB total), 256B aligned slices. All slices are
  // fully rewritten before being read on every call (deterministic).
  char* w = (char*)d_ws;
  auto carve = [&](size_t bytes) -> char* {
    char* p = w; w += (bytes + 255) & ~(size_t)255; return p;
  };
  float*  proj    = (float*)carve((size_t)NN * HID * 4);       // 256 MB
  float*  scores  = (float*)carve((size_t)NN * HEADS * 4);     //   2 MB
  float*  asum    = (float*)carve((size_t)NN * 4);             // 0.5 MB
  __bf16* Wh      = (__bf16*)carve((size_t)SEQ * HID * 2);     //   1 MB
  __bf16* Wl      = (__bf16*)carve((size_t)SEQ * HID * 2);     //   1 MB
  __bf16* A1h     = (__bf16*)carve((size_t)TWOH * DENSE * 2);  //  32 KB
  __bf16* A1l     = (__bf16*)carve((size_t)TWOH * DENSE * 2);  //  32 KB
  float*  segm    = (float*)carve((size_t)BNUM * HEADS * 4);
  float*  segd    = (float*)carve((size_t)BNUM * HEADS * 4);
  int*    segstart= (int*)carve((size_t)(BNUM + 1) * 4);
  float*  pooled  = (float*)carve((size_t)BNUM * TWOH * 4);
  float*  emb     = (float*)carve((size_t)BNUM * HID * 4);

  // 0) weight prep (tiny) + segment boundaries
  k_split_wproj<<<(SEQ * HID + 255) / 256, 256, 0, stream>>>(W_proj, Wh, Wl);
  k_split_wa1  <<<(TWOH * DENSE + 255) / 256, 256, 0, stream>>>(W_a1, A1h, A1l);
  k_bounds     <<<1, 128, 0, stream>>>(bid, segstart);

  // 1) dominant GEMM: proj = seq @ W_proj + b (bf16x3 WMMA, async-LDS staged)
  k_proj_gemm<<<NN / 32, 256, 0, stream>>>(seq, Wh, Wl, b_proj, proj);

  // 2) attention scores (WMMA + tanh + shuffle-reduced 16x4 matmul)
  k_scores<<<(NN / 16) / 8, 256, 0, stream>>>(structure, proj, A1h, A1l,
                                              b_a1, W_a2, b_a2, scores);

  // 3) per-graph softmax stats, then attn + per-node weight
  k_seg_stats<<<BNUM, 256, 0, stream>>>((const float4*)scores, segstart,
                                        (float4*)segm, (float4*)segd);
  k_attn<<<NN / 256, 256, 0, stream>>>((const float4*)scores, bid,
                                       (const float4*)segm, (const float4*)segd,
                                       (float4*)(out + BNUM * 2), asum);

  // 4) pooled segment-sum + final MLP head
  k_pooled<<<BNUM, 256, 0, stream>>>(structure, proj, asum, segstart, pooled);
  k_emb   <<<BNUM, 256, 0, stream>>>(pooled, W_f1, b_f1, emb);
  k_out   <<<1, 128, 0, stream>>>(emb, W_f2, b_f2, out);
}
